// TopKSAE_63745904607657
// MI455X (gfx1250) — compile-verified, matrix operand order checked by reading, not run on hardware
//
#include <hip/hip_runtime.h>
#include <hip/hip_bf16.h>

// ---------------- problem constants ----------------
#define NTOK   65536
#define IND    640          // input dim
#define LATD   2560         // latent dim
#define KTOP   32
#define MT     16           // tokens per block
#define BLOCK  256          // 8 waves of 32
#define NWAVE  8
#define NT_PER_WAVE (LATD / 16 / NWAVE)   // 20 n-tiles of 16 cols per wave
#define KCHUNKS     (IND / 32)            // 20 k-chunks of 32

// ---------------- LDS layout (dynamic) ----------------
#define XROW   650                        // padded bf16 row stride (conflict-free A loads)
#define LROW   2564                       // padded f32 latent row stride (conflict-free C stores)
#define SM_X     0
#define SM_LAT   (SM_X + MT * XROW * 2)                 // 20800  (16B aligned)
#define SM_LIST  (SM_LAT + MT * LROW * 4)               // 184896
#define SM_TOTAL (SM_LIST + MT * 48 * 4)                // 187968 bytes < 320KB WGP LDS

typedef __attribute__((ext_vector_type(16))) __bf16 v16bf;
typedef __attribute__((ext_vector_type(8)))  float  v8f;

union ABFrag {
    v16bf        v;
    unsigned int d[8];
};

__device__ __forceinline__ unsigned short f2bf(float f) {
    unsigned u = __float_as_uint(f);
    u += 0x7FFFu + ((u >> 16) & 1u);      // round-to-nearest-even
    return (unsigned short)(u >> 16);
}

// order-preserving float -> uint key (for exact top-k threshold bisection)
__device__ __forceinline__ unsigned fkey(float f) {
    unsigned u = __float_as_uint(f);
    unsigned m = (unsigned)((int)u >> 31) | 0x80000000u;
    return u ^ m;
}

// ---------------- pre-pass kernels ----------------
__global__ __launch_bounds__(256) void k_wenc_to_bf16(const float* __restrict__ w,
                                                      unsigned short* __restrict__ wb) {
    int i = blockIdx.x * 256 + threadIdx.x;       // grid sized exactly LATD*IND
    wb[i] = f2bf(w[i]);
}

__global__ __launch_bounds__(256) void k_wdec_transpose(const float* __restrict__ wd,
                                                        float* __restrict__ wdT) {
    int i = blockIdx.x * 256 + threadIdx.x;       // flat index into wdT [LATD][IND]
    int j = i / IND;                              // latent index
    int o = i % IND;                              // output index
    wdT[i] = wd[(size_t)o * LATD + j];            // W_dec is [IND][LATD]
}

// ---------------- fused encode + top-k + sparse decode ----------------
__global__ __launch_bounds__(BLOCK) void k_sae_fused(
        const float* __restrict__ x,
        const unsigned short* __restrict__ wencb,   // bf16 W_enc [LATD][IND]
        const float* __restrict__ benc,
        const float* __restrict__ wdT,              // f32 W_dec^T [LATD][IND]
        const float* __restrict__ bdec,
        float* __restrict__ recon,                  // [NTOK][IND]
        float* __restrict__ sparse)                 // [NTOK][LATD]
{
    extern __shared__ char smem[];
    unsigned short* xs   = (unsigned short*)(smem + SM_X);
    unsigned int*   xs32 = (unsigned int*)(smem + SM_X);
    float*          lat  = (float*)(smem + SM_LAT);
    int*            lst  = (int*)(smem + SM_LIST);

    const int tid  = threadIdx.x;
    const int wave = tid >> 5;
    const int lane = tid & 31;
    const int hi   = lane >> 4;
    const int ln   = lane & 15;
    const int t0   = blockIdx.x * MT;

    // ---- phase 1: stage x tile as bf16 into LDS (float4 loads, packed stores) ----
    {
        const float4* x4p = (const float4*)(x + (size_t)t0 * IND);
        for (int it = tid; it < MT * (IND / 4); it += BLOCK) {   // 10 iters/thread
            const int m  = it / (IND / 4);
            const int kq = it % (IND / 4);
            const float4 v = x4p[m * (IND / 4) + kq];
            const unsigned lo = (unsigned)f2bf(v.x) | ((unsigned)f2bf(v.y) << 16);
            const unsigned hh = (unsigned)f2bf(v.z) | ((unsigned)f2bf(v.w) << 16);
            const int di = m * (XROW / 2) + kq * 2;              // dword index
            xs32[di]     = lo;
            xs32[di + 1] = hh;
        }
    }
    __syncthreads();

    // ---- phase 2: bf16 WMMA GEMM, latents -> LDS ----
    for (int nt = 0; nt < NT_PER_WAVE; ++nt) {
        const int n0 = (wave * NT_PER_WAVE + nt) * 16;
        v8f acc = {0.f, 0.f, 0.f, 0.f, 0.f, 0.f, 0.f, 0.f};
        const unsigned short* brow = wencb + (size_t)(n0 + ln) * IND;
        __builtin_prefetch(brow, 0, 1);
        for (int kc = 0; kc < KCHUNKS; ++kc) {
            const int k0 = kc * 32;
            ABFrag a, b;
#pragma unroll
            for (int v = 0; v < 8; ++v) {
                // ISA 16-bit A/B fragment K mapping (pairs), lanes 16-31 offset by 8
                const int kb = ((v < 4) ? (2 * v) : (16 + 2 * (v - 4))) + 8 * hi;
                a.d[v] = *(const unsigned int*)&xs[ln * XROW + k0 + kb];
                b.d[v] = *(const unsigned int*)&brow[k0 + kb];
            }
            acc = __builtin_amdgcn_wmma_f32_16x16x32_bf16(
                      false, a.v, false, b.v, (short)0, acc, false, false);
        }
        const float be = benc[n0 + ln];
#pragma unroll
        for (int r = 0; r < 8; ++r) {
            const int m = r + 8 * hi;              // C/D row mapping
            lat[m * LROW + n0 + ln] = acc[r] + be;
        }
    }
    __syncthreads();

    // ---- phase 3: per-token exact top-32, sparse write, sparse decode ----
    const unsigned long long ltm = (1ull << lane) - 1ull;   // lanes below me
    for (int tt = 0; tt < 2; ++tt) {
        const int tok = wave + NWAVE * tt;         // 16 tokens / 8 waves
        const size_t t = (size_t)(t0 + tok);

        // cache this lane's 80 latent keys in VGPRs (element = q*128 + lane*4 + c)
        unsigned kreg[80];
#pragma unroll
        for (int q = 0; q < 20; ++q) {
            const float4 f4 = *(const float4*)&lat[tok * LROW + q * 128 + lane * 4];
            kreg[4 * q + 0] = fkey(f4.x);
            kreg[4 * q + 1] = fkey(f4.y);
            kreg[4 * q + 2] = fkey(f4.z);
            kreg[4 * q + 3] = fkey(f4.w);
        }

        // 32-step bisection: largest T with count(key >= T) >= KTOP == K-th key.
        // Wave-wide count via ballot+popc: v_cmp -> SGPR, count on scalar pipe
        // (co-issues with VALU) -- no shuffle butterfly needed on wave32.
        unsigned cur = 0;
        for (int bit = 31; bit >= 0; --bit) {
            const unsigned cand = cur | (1u << bit);
            int c = 0;
#pragma unroll
            for (int j = 0; j < 80; ++j)
                c += (int)__popcll(__ballot(kreg[j] >= cand));
            if (c >= KTOP) cur = cand;
        }

        // write sparse row (b128 stores) + ballot-prefix index compaction
        int base = 0;
        float* outS = sparse + t * LATD;
#pragma unroll
        for (int q = 0; q < 20; ++q) {
            const int e0 = q * 128 + lane * 4;
            const float4 f4 = *(const float4*)&lat[tok * LROW + e0];
            float4 o4;
            {
                const bool s = kreg[4 * q + 0] >= cur;
                const unsigned long long m = __ballot(s);
                o4.x = s ? f4.x : 0.0f;
                if (s) { int p = base + __popcll(m & ltm); if (p < 48) lst[tok * 48 + p] = e0 + 0; }
                base += __popcll(m);
            }
            {
                const bool s = kreg[4 * q + 1] >= cur;
                const unsigned long long m = __ballot(s);
                o4.y = s ? f4.y : 0.0f;
                if (s) { int p = base + __popcll(m & ltm); if (p < 48) lst[tok * 48 + p] = e0 + 1; }
                base += __popcll(m);
            }
            {
                const bool s = kreg[4 * q + 2] >= cur;
                const unsigned long long m = __ballot(s);
                o4.z = s ? f4.z : 0.0f;
                if (s) { int p = base + __popcll(m & ltm); if (p < 48) lst[tok * 48 + p] = e0 + 2; }
                base += __popcll(m);
            }
            {
                const bool s = kreg[4 * q + 3] >= cur;
                const unsigned long long m = __ballot(s);
                o4.w = s ? f4.w : 0.0f;
                if (s) { int p = base + __popcll(m & ltm); if (p < 48) lst[tok * 48 + p] = e0 + 3; }
                base += __popcll(m);
            }
            *(float4*)&outS[e0] = o4;
        }
        __threadfence_block();

        // sparse decode: recon = b_dec + sum_k val_k * W_dec^T[idx_k, :]
        const int cn = base < 48 ? base : 48;
        float4 acc4[5];
        const float4* bd4 = (const float4*)bdec;
#pragma unroll
        for (int i4 = 0; i4 < 5; ++i4) acc4[i4] = bd4[i4 * 32 + lane];
        for (int e = 0; e < cn; ++e) {
            const int   idx = lst[tok * 48 + e];
            const float val = lat[tok * LROW + idx];
            const float4* w4 = (const float4*)(wdT + (size_t)idx * IND);
#pragma unroll
            for (int i4 = 0; i4 < 5; ++i4) {
                const float4 w = w4[i4 * 32 + lane];
                acc4[i4].x += val * w.x;
                acc4[i4].y += val * w.y;
                acc4[i4].z += val * w.z;
                acc4[i4].w += val * w.w;
            }
        }
        float4* outR4 = (float4*)(recon + t * IND);
#pragma unroll
        for (int i4 = 0; i4 < 5; ++i4) outR4[i4 * 32 + lane] = acc4[i4];
    }
}

// ---------------- launcher ----------------
extern "C" void kernel_launch(void* const* d_in, const int* in_sizes, int n_in,
                              void* d_out, int out_size, void* d_ws, size_t ws_size,
                              hipStream_t stream) {
    (void)in_sizes; (void)n_in; (void)out_size; (void)ws_size;
    const float* x     = (const float*)d_in[0];
    const float* W_enc = (const float*)d_in[1];
    const float* b_enc = (const float*)d_in[2];
    const float* W_dec = (const float*)d_in[3];
    const float* b_dec = (const float*)d_in[4];

    float* recon  = (float*)d_out;                          // [NTOK][IND]
    float* sparse = recon + (size_t)NTOK * IND;             // [NTOK][LATD]

    unsigned short* wencb = (unsigned short*)d_ws;          // 3,276,800 B
    float* wdT = (float*)((char*)d_ws + (size_t)LATD * IND * 2); // already 256B aligned

    const int nW = (LATD * IND) / 256;                      // 6400 blocks
    k_wenc_to_bf16 <<<nW, 256, 0, stream>>>(W_enc, wencb);
    k_wdec_transpose<<<nW, 256, 0, stream>>>(W_dec, wdT);

    k_sae_fused<<<NTOK / MT, BLOCK, SM_TOTAL, stream>>>(
        x, wencb, b_enc, wdT, b_dec, recon, sparse);
}
